// TinyGRU_36979668418798
// MI455X (gfx1250) — matrix operand order repaired and checked by reading, not compiled, not run on hardware
//
#include <hip/hip_runtime.h>

// ---------------------------------------------------------------------------
// TinyGRU on MI455X (gfx1250): bf16 WMMA GEMMs, f32 accumulate.
// All GEMM kernels: TDM (tensor_load_to_lds) double-buffers the activation
// slab into LDS (shared by 8 waves/block); weights stream from global/L2.
// L=8 layers, T=8 steps, B=512, H=O=IN=1024.
// ---------------------------------------------------------------------------

typedef __attribute__((ext_vector_type(16))) __bf16        v16bf;
typedef __attribute__((ext_vector_type(8)))  float         v8f;
typedef __attribute__((ext_vector_type(4)))  unsigned int  v4u;
typedef __attribute__((ext_vector_type(8)))  int           v8i;
typedef __attribute__((ext_vector_type(4)))  int           v4i;

union frag_u { v16bf bf; v4u u4[2]; };

#define HH 1024
#define BB 512
#define LL 8
#define TT 8
#define OO 1024

#if __has_builtin(__builtin_amdgcn_tensor_load_to_lds)
#define HAVE_TDM 1
#else
#define HAVE_TDM 0
#endif

__device__ __forceinline__ unsigned short f32_to_bf16(float f) {
  union { float f; unsigned int u; } v; v.f = f;
  unsigned int u = v.u;
  u += 0x7fffu + ((u >> 16) & 1u);     // round-to-nearest-even
  return (unsigned short)(u >> 16);
}

__device__ __forceinline__ float sigmoidf_(float x) {
  return 1.0f / (1.0f + __expf(-x));
}

// A-fragment layout (16-bit A 16x32, ISA 7.12.2): lane holds row M=lane&15;
// lanes 0-15: K = 0..7 and 16..23 ; lanes 16-31: K = 8..15 and 24..31.
// Slab resident in LDS as [row][32 K] (row length 32).
__device__ __forceinline__ void load_a_lds(frag_u& f, const unsigned short* S,
                                           int lane) {
  const int half = lane >> 4, row = lane & 15;
  const unsigned short* p = S + row * 32 + half * 8;
  f.u4[0] = *(const v4u*)(p);
  f.u4[1] = *(const v4u*)(p + 16);
}

// B-fragment (32x16, 16-bit): lane holds column N=lane&15;
// lanes 0-15: K = 0..7 and 16..23 ; lanes 16-31: K = 8..15 and 24..31.  Bt is the TRANSPOSED weight: Bt[n][k].
__device__ __forceinline__ void load_b(frag_u& f, const unsigned short* Bt,
                                       int n0, int kk, int lane) {
  const int half = lane >> 4, col = lane & 15;
  const unsigned short* p = Bt + (size_t)(n0 + col) * HH + kk + half * 8;
  f.u4[0] = *(const v4u*)(p);
  f.u4[1] = *(const v4u*)(p + 16);
}

__device__ __forceinline__ v8f wmma_bf16(const frag_u& a, const frag_u& b, v8f c) {
  return __builtin_amdgcn_wmma_f32_16x16x32_bf16(false, a.bf, false, b.bf,
                                                 (short)0, c, false, false);
}

#if HAVE_TDM
// TDM: DMA a rows x 32 (bf16) tile from global (row stride ld_elems) into LDS.
// D# packing per CDNA5 ISA 8.3/8.4: group0 = {flags, lds_addr, global_addr,
// type=2}; group1 = {data_size=2B, tensor dims, tile dims, dim0 stride}.
// 6-arg builtin on this toolchain: (g0, g1, g2, g3, g4, cpol); unused groups
// (2D tensor) zero-filled.
__device__ __forceinline__ void tdm_load_2d(unsigned lds_off,
                                            const unsigned short* gptr,
                                            int rows, int ld_elems) {
  const unsigned long long ga = (unsigned long long)(size_t)gptr;
  v4u g0;
  g0[0] = 1u;                                       // count=1, user descriptor
  g0[1] = lds_off;                                  // lds_addr
  g0[2] = (unsigned)ga;                             // global_addr[31:0]
  g0[3] = (unsigned)((ga >> 32) & 0x1FFFFFFu) | (2u << 30);  // ga[56:32]|type=2
  v8i g1;
  g1[0] = (int)(1u << 16);                          // data_size = 2 bytes
  g1[1] = (int)(((unsigned)ld_elems) << 16);        // tensor_dim0[15:0]
  g1[2] = (int)((((unsigned)ld_elems) >> 16) | (((unsigned)rows) << 16));
  g1[3] = (int)((((unsigned)rows) >> 16) | (32u << 16));     // tile_dim0 = 32
  g1[4] = (int)(unsigned)rows;                      // tile_dim1 = rows
  g1[5] = (int)(unsigned)ld_elems;                  // tensor_dim0_stride lo32
  g1[6] = 0;                                        // stride hi / dim1 stride
  g1[7] = 0;
  const v4i z4 = {0, 0, 0, 0};
  const v8i z8 = {0, 0, 0, 0, 0, 0, 0, 0};
  __builtin_amdgcn_tensor_load_to_lds(g0, g1, z4, z4, z8, 0);
}
#endif

// ---------------------------------------------------------------------------
// Weight convert + transpose: dst[n*H+k] = bf16(src[k*H+n]); LDS tiled.
// ---------------------------------------------------------------------------
__global__ __launch_bounds__(256)
void k_convert_transpose(const float* __restrict__ src, unsigned short* __restrict__ dst) {
  __shared__ float tile[32][33];
  const int x  = blockIdx.x * 32 + threadIdx.x;
  const int y0 = blockIdx.y * 32;
  for (int j = 0; j < 32; j += 8)
    tile[threadIdx.y + j][threadIdx.x] = src[(size_t)(y0 + threadIdx.y + j) * HH + x];
  __syncthreads();
  const int yo = blockIdx.x * 32;
  const int xo = blockIdx.y * 32 + threadIdx.x;
  for (int j = 0; j < 32; j += 8)
    dst[(size_t)(yo + threadIdx.y + j) * HH + xo] =
        f32_to_bf16(tile[threadIdx.x][threadIdx.y + j]);
}

// x (B,S=128,IN) f32 -> xb (T,B,IN) bf16, first T timesteps only
__global__ __launch_bounds__(256)
void k_convert_x(const float* __restrict__ x, unsigned short* __restrict__ xb) {
  const size_t idx = (size_t)blockIdx.x * blockDim.x + threadIdx.x;
  const int i = idx % HH;
  const size_t rest = idx / HH;
  const int b = rest % BB;
  const int t = rest / BB;
  xb[((size_t)t * BB + b) * HH + i] = f32_to_bf16(x[((size_t)b * 128 + t) * HH + i]);
}

__global__ __launch_bounds__(256)
void k_init_h(const float* __restrict__ hp, float* __restrict__ hf,
              unsigned short* __restrict__ hb) {
  const size_t idx = (size_t)blockIdx.x * blockDim.x + threadIdx.x;
  const float v = hp[idx];
  hf[idx] = v;
  hb[idx] = f32_to_bf16(v);
}

// ---------------------------------------------------------------------------
// Gates: block = 8 waves (2 M-waves x 4 N-waves), block tile 32(M) x 128(N).
// A slabs (inp, hp: 32x32 bf16 each) double-buffered in LDS via TDM;
// each wave owns 2 N-tiles -> 6 f32 accumulators, 10 WMMAs per K-step.
// ---------------------------------------------------------------------------
__global__ __launch_bounds__(256)
void k_gru_gates(const unsigned short* __restrict__ inp_b,
                 const unsigned short* __restrict__ hp_b,
                 const float* __restrict__ hp_f,
                 const unsigned short* __restrict__ Wxz,
                 const unsigned short* __restrict__ Whz,
                 const unsigned short* __restrict__ Wxr,
                 const unsigned short* __restrict__ Whr,
                 const unsigned short* __restrict__ Wxh,
                 const float* __restrict__ bz, const float* __restrict__ br,
                 const float* __restrict__ bh,
                 float* __restrict__ z_o, float* __restrict__ r_o,
                 float* __restrict__ t_o, unsigned short* __restrict__ rh_b) {
  __shared__ unsigned short sA[2][2][32][32];   // [buf][inp|hp][row][k], 8 KB
  const int lane = threadIdx.x & 31;
  const int wave = threadIdx.x >> 5;
  const int wm   = wave >> 2;                   // 0..1
  const int wn   = wave & 3;                    // 0..3
  const int mblk = blockIdx.x * 32;             // grid.x = B/32 = 16
  const int m0   = mblk + wm * 16;
  const int n0   = blockIdx.y * 128 + wn * 32;  // grid.y = H/128 = 8

#if HAVE_TDM
  if (wave == 0) {
    tdm_load_2d((unsigned)(size_t)&sA[0][0][0][0], inp_b + (size_t)mblk * HH, 32, HH);
    tdm_load_2d((unsigned)(size_t)&sA[0][1][0][0], hp_b  + (size_t)mblk * HH, 32, HH);
  }
#endif

  v8f az[2] = {{}, {}}, ar[2] = {{}, {}}, at[2] = {{}, {}};

  int buf = 0;
  for (int kk = 0; kk < HH; kk += 32) {
#if HAVE_TDM
    if (wave == 0) __builtin_amdgcn_s_wait_tensorcnt(0);   // current buf landed
    __syncthreads();                                       // visible to all
    if (wave == 0 && kk + 32 < HH) {                       // prefetch next slab
      tdm_load_2d((unsigned)(size_t)&sA[buf ^ 1][0][0][0],
                  inp_b + (size_t)mblk * HH + kk + 32, 32, HH);
      tdm_load_2d((unsigned)(size_t)&sA[buf ^ 1][1][0][0],
                  hp_b + (size_t)mblk * HH + kk + 32, 32, HH);
    }
#else
    __syncthreads();                       // previous compute done
    {                                      // 256 threads x 16B = both slabs
      const int mat = threadIdx.x >> 7, row = (threadIdx.x >> 2) & 31,
                kg = threadIdx.x & 3;
      const unsigned short* src = (mat ? hp_b : inp_b) +
          (size_t)(mblk + row) * HH + kk + kg * 8;
      *(v4u*)&sA[0][mat][row][kg * 8] = *(const v4u*)src;
    }
    __syncthreads();
#endif

    frag_u ai, ah;
    load_a_lds(ai, &sA[buf][0][wm * 16][0], lane);
    load_a_lds(ah, &sA[buf][1][wm * 16][0], lane);

    #pragma unroll
    for (int nt = 0; nt < 2; ++nt) {
      const int n = n0 + nt * 16;
      frag_u b0, b1, b2, b3, b4;
      load_b(b0, Wxz, n, kk, lane);
      load_b(b1, Whz, n, kk, lane);
      load_b(b2, Wxr, n, kk, lane);
      load_b(b3, Whr, n, kk, lane);
      load_b(b4, Wxh, n, kk, lane);
      az[nt] = wmma_bf16(ai, b0, az[nt]);
      az[nt] = wmma_bf16(ah, b1, az[nt]);
      ar[nt] = wmma_bf16(ai, b2, ar[nt]);
      ar[nt] = wmma_bf16(ah, b3, ar[nt]);
      at[nt] = wmma_bf16(ai, b4, at[nt]);
    }
#if HAVE_TDM
    buf ^= 1;
#endif
  }

  const int half = lane >> 4;
  #pragma unroll
  for (int nt = 0; nt < 2; ++nt) {
    const int col = n0 + nt * 16 + (lane & 15);
    const float bzv = bz[col], brv = br[col], bhv = bh[col];
    #pragma unroll
    for (int i = 0; i < 8; ++i) {
      const int row = m0 + half * 8 + i;
      const size_t off = (size_t)row * HH + col;
      const float zf  = sigmoidf_(az[nt][i] + bzv);
      const float rf  = sigmoidf_(ar[nt][i] + brv);
      const float hpv = hp_f[off];
      z_o[off]  = zf;
      r_o[off]  = rf;
      t_o[off]  = at[nt][i] + bhv;
      rh_b[off] = f32_to_bf16(rf * hpv);
    }
  }
}

// ---------------------------------------------------------------------------
// Update: h_tilde = tanh(tmp + rh@Whh); h = (1-z)*hp + z*h_tilde.
// Same TDM/LDS block structure: A slab (rh) staged once per block.
// ---------------------------------------------------------------------------
__global__ __launch_bounds__(256)
void k_gru_update(const unsigned short* __restrict__ rh_b,
                  const unsigned short* __restrict__ Whh,
                  const float* __restrict__ t_i, const float* __restrict__ z_i,
                  float* __restrict__ h_f, unsigned short* __restrict__ h_b,
                  float* __restrict__ hseq, float* __restrict__ hcur) {
  __shared__ unsigned short sA[2][32][32];      // [buf][row][k], 4 KB
  const int lane = threadIdx.x & 31;
  const int wave = threadIdx.x >> 5;
  const int wm   = wave >> 2;
  const int wn   = wave & 3;
  const int mblk = blockIdx.x * 32;             // grid (16, 8)
  const int m0   = mblk + wm * 16;
  const int n0   = blockIdx.y * 128 + wn * 32;

#if HAVE_TDM
  if (wave == 0)
    tdm_load_2d((unsigned)(size_t)&sA[0][0][0], rh_b + (size_t)mblk * HH, 32, HH);
#endif

  v8f acc[2] = {{}, {}};
  int buf = 0;
  for (int kk = 0; kk < HH; kk += 32) {
#if HAVE_TDM
    if (wave == 0) __builtin_amdgcn_s_wait_tensorcnt(0);
    __syncthreads();
    if (wave == 0 && kk + 32 < HH)
      tdm_load_2d((unsigned)(size_t)&sA[buf ^ 1][0][0],
                  rh_b + (size_t)mblk * HH + kk + 32, 32, HH);
#else
    __syncthreads();
    if (threadIdx.x < 128) {                   // 128 threads x 16B = slab
      const int row = threadIdx.x >> 2, kg = threadIdx.x & 3;
      *(v4u*)&sA[0][row][kg * 8] =
          *(const v4u*)(rh_b + (size_t)(mblk + row) * HH + kk + kg * 8);
    }
    __syncthreads();
#endif

    frag_u a;
    load_a_lds(a, &sA[buf][wm * 16][0], lane);
    if (kk + 32 < HH)
      __builtin_prefetch(Whh + (size_t)(n0 + (lane & 15)) * HH + kk + 32, 0, 1);
    #pragma unroll
    for (int nt = 0; nt < 2; ++nt) {
      frag_u b;
      load_b(b, Whh, n0 + nt * 16, kk, lane);
      acc[nt] = wmma_bf16(a, b, acc[nt]);
    }
#if HAVE_TDM
    buf ^= 1;
#endif
  }

  const int half = lane >> 4;
  #pragma unroll
  for (int nt = 0; nt < 2; ++nt) {
    const int col = n0 + nt * 16 + (lane & 15);
    #pragma unroll
    for (int i = 0; i < 8; ++i) {
      const int row = m0 + half * 8 + i;
      const size_t off = (size_t)row * HH + col;
      const float ht  = tanhf(t_i[off] + acc[nt][i]);
      const float zf  = z_i[off];
      const float hpv = h_f[off];
      const float hn  = (1.0f - zf) * hpv + zf * ht;
      h_f[off]  = hn;
      h_b[off]  = f32_to_bf16(hn);
      hseq[off] = hn;
      hcur[off] = hn;
    }
  }
}

// ---------------------------------------------------------------------------
// Output projection: y[b, t, :] = h7 @ W_hy + b_y  (row stride T*O).
// Same TDM/LDS block structure.
// ---------------------------------------------------------------------------
__global__ __launch_bounds__(256)
void k_out_proj(const unsigned short* __restrict__ hb,
                const unsigned short* __restrict__ Why,
                const float* __restrict__ by, float* __restrict__ y, int ldy) {
  __shared__ unsigned short sA[2][32][32];
  const int lane = threadIdx.x & 31;
  const int wave = threadIdx.x >> 5;
  const int wm   = wave >> 2;
  const int wn   = wave & 3;
  const int mblk = blockIdx.x * 32;             // grid (16, 8)
  const int m0   = mblk + wm * 16;
  const int n0   = blockIdx.y * 128 + wn * 32;

#if HAVE_TDM
  if (wave == 0)
    tdm_load_2d((unsigned)(size_t)&sA[0][0][0], hb + (size_t)mblk * HH, 32, HH);
#endif

  v8f acc[2] = {{}, {}};
  int buf = 0;
  for (int kk = 0; kk < HH; kk += 32) {
#if HAVE_TDM
    if (wave == 0) __builtin_amdgcn_s_wait_tensorcnt(0);
    __syncthreads();
    if (wave == 0 && kk + 32 < HH)
      tdm_load_2d((unsigned)(size_t)&sA[buf ^ 1][0][0],
                  hb + (size_t)mblk * HH + kk + 32, 32, HH);
#else
    __syncthreads();
    if (threadIdx.x < 128) {
      const int row = threadIdx.x >> 2, kg = threadIdx.x & 3;
      *(v4u*)&sA[0][row][kg * 8] =
          *(const v4u*)(hb + (size_t)(mblk + row) * HH + kk + kg * 8);
    }
    __syncthreads();
#endif

    frag_u a;
    load_a_lds(a, &sA[buf][wm * 16][0], lane);
    #pragma unroll
    for (int nt = 0; nt < 2; ++nt) {
      frag_u b;
      load_b(b, Why, n0 + nt * 16, kk, lane);
      acc[nt] = wmma_bf16(a, b, acc[nt]);
    }
#if HAVE_TDM
    buf ^= 1;
#endif
  }

  const int half = lane >> 4;
  #pragma unroll
  for (int nt = 0; nt < 2; ++nt) {
    const int col = n0 + nt * 16 + (lane & 15);
    const float byv = by[col];
    #pragma unroll
    for (int i = 0; i < 8; ++i) {
      const int row = m0 + half * 8 + i;
      y[(size_t)row * ldy + col] = acc[nt][i] + byv;
    }
  }
}

// ---------------------------------------------------------------------------
// Host orchestration
// ---------------------------------------------------------------------------
extern "C" void kernel_launch(void* const* d_in, const int* in_sizes, int n_in,
                              void* d_out, int out_size, void* d_ws, size_t ws_size,
                              hipStream_t stream) {
  (void)in_sizes; (void)n_in; (void)out_size; (void)ws_size;

  const float* x      = (const float*)d_in[0];
  const float* h_prev = (const float*)d_in[1];
  const float* W_xz   = (const float*)d_in[2];
  const float* W_hz   = (const float*)d_in[3];
  const float* b_z    = (const float*)d_in[4];
  const float* W_xr   = (const float*)d_in[5];
  const float* W_hr   = (const float*)d_in[6];
  const float* b_r    = (const float*)d_in[7];
  const float* W_xh   = (const float*)d_in[8];
  const float* W_hh   = (const float*)d_in[9];
  const float* b_h    = (const float*)d_in[10];
  const float* W_hy   = (const float*)d_in[11];
  const float* b_y    = (const float*)d_in[12];

  const size_t MAT = (size_t)HH * HH;
  const size_t BH  = (size_t)BB * HH;

  // ---- workspace layout ----
  unsigned short* Wt  = (unsigned short*)d_ws;          // 49 * MAT bf16 (transposed)
  unsigned short* xb  = Wt + 49 * MAT;                  // T*B*H bf16
  unsigned short* hb  = xb + (size_t)TT * BH;           // L*B*H bf16 state shadow
  unsigned short* rhb = hb + (size_t)LL * BH;           // B*H bf16 (r*hp)
  float* hf = (float*)(rhb + BH);                       // L*B*H f32 state
  float* zb = hf + (size_t)LL * BH;
  float* rb = zb + BH;
  float* tb = rb + BH;

  // ---- output layout: outputs (B,T,O) | h_seq (L,T,B,H) | h_current (L,B,H)
  float* outputs = (float*)d_out;
  float* hseq    = outputs + (size_t)BB * TT * OO;
  float* hcur    = hseq + (size_t)LL * TT * BH;

  // ---- one-time converts ----
  const dim3 tgrid(HH / 32, HH / 32), tblk(32, 8);
  const float* srcs[6] = { W_xz, W_hz, W_xr, W_hr, W_xh, W_hh };
  for (int l = 0; l < LL; ++l)
    for (int m = 0; m < 6; ++m)
      k_convert_transpose<<<tgrid, tblk, 0, stream>>>(srcs[m] + (size_t)l * MAT,
                                                      Wt + ((size_t)l * 6 + m) * MAT);
  k_convert_transpose<<<tgrid, tblk, 0, stream>>>(W_hy, Wt + 48 * MAT);
  k_convert_x<<<(int)((size_t)TT * BH / 256), 256, 0, stream>>>(x, xb);
  k_init_h<<<(int)((size_t)LL * BH / 256), 256, 0, stream>>>(h_prev, hf, hb);

  // ---- sequential GRU ----
  const dim3 ggrid(BB / 32, HH / 128);   // (16, 8) blocks of 8 waves
  for (int t = 0; t < TT; ++t) {
    const unsigned short* inp = xb + (size_t)t * BH;
    for (int l = 0; l < LL; ++l) {
      unsigned short* hb_l = hb + (size_t)l * BH;
      float*          hf_l = hf + (size_t)l * BH;
      const unsigned short* Wl = Wt + (size_t)l * 6 * MAT;
      k_gru_gates<<<ggrid, 256, 0, stream>>>(
          inp, hb_l, hf_l,
          Wl + 0 * MAT, Wl + 1 * MAT, Wl + 2 * MAT, Wl + 3 * MAT, Wl + 4 * MAT,
          b_z + (size_t)l * HH, b_r + (size_t)l * HH, b_h + (size_t)l * HH,
          zb, rb, tb, rhb);
      k_gru_update<<<ggrid, 256, 0, stream>>>(
          rhb, Wl + 5 * MAT, tb, zb, hf_l, hb_l,
          hseq + ((size_t)l * TT + t) * BH, hcur + (size_t)l * BH);
      inp = hb_l;
    }
    k_out_proj<<<ggrid, 256, 0, stream>>>(hb + (size_t)(LL - 1) * BH, Wt + 48 * MAT,
                                          b_y, outputs + (size_t)t * OO, TT * OO);
  }
}
